// SparseAttention_2619930051646
// MI455X (gfx1250) — matrix operand order repaired and checked by reading, not compiled, not run on hardware
//
#include <hip/hip_runtime.h>
#include <hip/hip_bf16.h>

typedef __attribute__((ext_vector_type(16))) _Float16 v16h;
typedef __attribute__((ext_vector_type(8)))  _Float16 v8h;
typedef __attribute__((ext_vector_type(8)))  float    v8f;
typedef __attribute__((ext_vector_type(4)))  float    v4f;

#define SB 4
#define SS 2048
#define SH 256
#define NH 4
#define HD 64
#define TOPK 10

// ---------------------------------------------------------------------------
// Kernel 1: zero-fill the dense attn output (67.1M floats = 268 MB).
// This is the roofline-dominant write, so use non-temporal b128 stores
// (keep the 192MB L2 for Q/K/V + score tiles) and 4 stores per thread.
// ---------------------------------------------------------------------------
__global__ __launch_bounds__(256) void zero_attn_kernel(v4f* __restrict__ p) {
    const size_t base = ((size_t)blockIdx.x * 256 + threadIdx.x) * 4;
    const v4f z = {0.f, 0.f, 0.f, 0.f};
#pragma unroll
    for (int i = 0; i < 4; ++i)
        __builtin_nontemporal_store(z, p + base + i);
}

// ---------------------------------------------------------------------------
// Kernel 2: fused QKV projection  y = x @ W^T + bias  via WMMA f16 (f32 acc).
// One wave computes one 16x16 output tile; K=256 in 8 chunks of 32.
// A-frag (16x32 f16): lane m=lane&15, g=lane>>4; elems j<8 -> K = kc+g*8+j,
// elems j>=8 -> K = kc+16+g*8+(j-8)   (ISA 16-bit A-matrix table).
// B-frag (32x16 f16): lane n=lane&15, g=lane>>4; same K mapping as A-frag,
// column n = row n of W (since B = W^T restricted to tile).
// Q,K stored f16 [b,h,s,d]; V stored f32 [b,h,s,d].
// ---------------------------------------------------------------------------
__global__ __launch_bounds__(256) void qkv_proj_kernel(
    const float* __restrict__ x,
    const float* __restrict__ Wq, const float* __restrict__ Wk, const float* __restrict__ Wv,
    const float* __restrict__ bq, const float* __restrict__ bk, const float* __restrict__ bv,
    _Float16* __restrict__ Qh, _Float16* __restrict__ Kh, float* __restrict__ Vf)
{
    const int lane  = threadIdx.x & 31;
    const int wave  = threadIdx.x >> 5;
    const int mtile = blockIdx.x * 8 + wave;   // 0..511  (B*S/16)
    const int ntile = blockIdx.y;              // 0..15   (H/16)
    const int which = blockIdx.z;              // 0=Q 1=K 2=V

    const float* W    = (which == 0) ? Wq : (which == 1) ? Wk : Wv;
    const float* bias = (which == 0) ? bq : (which == 1) ? bk : bv;

    const int m = lane & 15;
    const int g = lane >> 4;
    const float* xrow = x + (size_t)(mtile * 16 + m) * SH;
    const float* wrow = W + (size_t)(ntile * 16 + m) * SH;   // lane's B-column n == m

    v8f c = {};
#pragma unroll 2
    for (int kc = 0; kc < SH; kc += 32) {
        v16h a, bf;
        const float* ax0 = xrow + kc + g * 8;
        const float* ax1 = xrow + kc + 16 + g * 8;
        const float* bx0 = wrow + kc + g * 8;
        const float* bx1 = wrow + kc + 16 + g * 8;
#pragma unroll
        for (int j = 0; j < 8; ++j) {
            a[j]     = (_Float16)ax0[j];
            a[8 + j] = (_Float16)ax1[j];
        }
#pragma unroll
        for (int j = 0; j < 8; ++j) { bf[j] = (_Float16)bx0[j]; bf[8 + j] = (_Float16)bx1[j]; }
        c = __builtin_amdgcn_wmma_f32_16x16x32_f16(false, a, false, bf,
                                                   (short)0, c, false, false);
    }

    const int n  = ntile * 16 + m;
    const float bn = bias[n];
    const int hh = n >> 6;      // head
    const int d  = n & 63;      // dim in head

    size_t offs[8];
    float  vals[8];
#pragma unroll
    for (int r = 0; r < 8; ++r) {
        const int mg = mtile * 16 + r + 8 * g;     // C layout: M = r + 8*g
        const int bb = mg >> 11;                   // batch
        const int s  = mg & 2047;                  // seq
        vals[r] = c[r] + bn;
        offs[r] = ((size_t)(bb * NH + hh) * SS + s) * HD + d;
    }
    if (which == 2) {
#pragma unroll
        for (int r = 0; r < 8; ++r) Vf[offs[r]] = vals[r];
    } else {
        _Float16* __restrict__ dst = (which == 0) ? Qh : Kh;
#pragma unroll
        for (int r = 0; r < 8; ++r) dst[offs[r]] = (_Float16)vals[r];
    }
}

// ---------------------------------------------------------------------------
// Kernel 3: per (b,h,16-query block): scores via WMMA, rolling top-10 per row,
// softmax over the 10 survivors, scatter into attn, sparse context = p * V.
// Block = 256 threads = 8 waves; each wave sweeps 2 key tiles per 256-key
// chunk; 16 threads merge the chunk into LDS top-10 lists.
// ---------------------------------------------------------------------------
__global__ __launch_bounds__(256) void sparse_attn_kernel(
    const _Float16* __restrict__ Qh, const _Float16* __restrict__ Kh,
    const float* __restrict__ Vf,
    float* __restrict__ attn, float* __restrict__ ctx)
{
    __shared__ float sc[16][264];          // 16 rows x 256-key chunk (+pad)
    __shared__ float topv[16][TOPK];
    __shared__ int   topi[16][TOPK];

    const int tid  = threadIdx.x;
    const int lane = tid & 31;
    const int wave = tid >> 5;
    const int qblock = blockIdx.x;   // 0..127
    const int h      = blockIdx.y;   // 0..3
    const int b      = blockIdx.z;   // 0..3
    const int bh     = b * NH + h;
    const size_t bhbase = (size_t)bh * SS;

    const int m = lane & 15;
    const int g = lane >> 4;

    // Q A-fragments, loaded once per wave (d-chunks 0..31 and 32..63)
    const v8h* qp = (const v8h*)(Qh + (bhbase + qblock * 16 + m) * HD);
    const v8h q0l = qp[2 * g],     q0h = qp[2 * g + 1];
    const v8h q1l = qp[4 + 2 * g], q1h = qp[5 + 2 * g];
    const v16h a0 = __builtin_shufflevector(q0l, q0h, 0,1,2,3,4,5,6,7,8,9,10,11,12,13,14,15);
    const v16h a1 = __builtin_shufflevector(q1l, q1h, 0,1,2,3,4,5,6,7,8,9,10,11,12,13,14,15);

    if (tid < 16) {
#pragma unroll
        for (int i = 0; i < TOPK; ++i) { topv[tid][i] = -3.0e38f; topi[tid][i] = 0; }
    }
    __syncthreads();

    for (int chunk = 0; chunk < 8; ++chunk) {
#pragma unroll
        for (int t = 0; t < 2; ++t) {
            const int tic = wave * 2 + t;              // tile index in chunk
            const int kt  = chunk * 16 + tic;          // key tile 0..127
            const _Float16* krow = Kh + (bhbase + kt * 16 + m) * HD;
            // prefetch this wave's K tile of the NEXT chunk into caches
            if (kt + 16 < 128)
                __builtin_prefetch((const void*)(krow + 16 * 16 * HD), 0, 0);
            const v16h b0 = *(const v16h*)(krow + g * 16);
            const v16h b1 = *(const v16h*)(krow + 32 + g * 16);
            v8f c = {};
            c = __builtin_amdgcn_wmma_f32_16x16x32_f16(false, a0, false, b0,
                                                       (short)0, c, false, false);
            c = __builtin_amdgcn_wmma_f32_16x16x32_f16(false, a1, false, b1,
                                                       (short)0, c, false, false);
            const int col = tic * 16 + m;              // key column within chunk
#pragma unroll
            for (int r = 0; r < 8; ++r)
                sc[r + 8 * g][col] = c[r] * 0.125f;    // /sqrt(64)
        }
        __syncthreads();

        if (tid < 16) {                                // per-row top-10 merge
            const int kb = chunk * 256;
            for (int j = 0; j < 256; ++j) {
                const float v = sc[tid][j];
                if (v > topv[tid][TOPK - 1]) {
                    int p = TOPK - 1;
                    while (p > 0 && v > topv[tid][p - 1]) {
                        topv[tid][p] = topv[tid][p - 1];
                        topi[tid][p] = topi[tid][p - 1];
                        --p;
                    }
                    topv[tid][p] = v;
                    topi[tid][p] = kb + j;
                }
            }
        }
        __syncthreads();
    }

    // softmax over the 10 survivors; scatter nonzeros into attn
    if (tid < 16) {
        const float mx = topv[tid][0];
        float s = 0.f;
#pragma unroll
        for (int i = 0; i < TOPK; ++i) {
            const float e = expf(topv[tid][i] - mx);
            topv[tid][i] = e;
            s += e;
        }
        const float inv = 1.f / s;
        const size_t rowoff = (bhbase + qblock * 16 + tid) * (size_t)SS;
#pragma unroll
        for (int i = 0; i < TOPK; ++i) {
            const float p = topv[tid][i] * inv;
            topv[tid][i] = p;
            attn[rowoff + topi[tid][i]] = p;
        }
    }
    __syncthreads();

    // sparse context: ctx[row, d] = sum_i p_i * V[k_i, d]
    const int row = tid >> 4;
    const int dq  = (tid & 15) * 4;
    float4 acc = make_float4(0.f, 0.f, 0.f, 0.f);
#pragma unroll
    for (int i = 0; i < TOPK; ++i) {
        const int   k = topi[row][i];
        const float p = topv[row][i];
        const float4 vv = *(const float4*)(Vf + (bhbase + k) * HD + dq);
        acc.x += p * vv.x; acc.y += p * vv.y; acc.z += p * vv.z; acc.w += p * vv.w;
    }
    const int q = qblock * 16 + row;
    *(float4*)(ctx + ((size_t)b * SS + q) * SH + h * HD + dq) = acc;
}

// ---------------------------------------------------------------------------
extern "C" void kernel_launch(void* const* d_in, const int* in_sizes, int n_in,
                              void* d_out, int out_size, void* d_ws, size_t ws_size,
                              hipStream_t stream) {
    (void)in_sizes; (void)n_in; (void)out_size; (void)ws_size;
    const float* x  = (const float*)d_in[0];
    const float* Wq = (const float*)d_in[1];
    const float* bq = (const float*)d_in[2];
    const float* Wk = (const float*)d_in[3];
    const float* bk = (const float*)d_in[4];
    const float* Wv = (const float*)d_in[5];
    const float* bv = (const float*)d_in[6];

    float* ctx  = (float*)d_out;                         // [4,2048,256]
    float* attn = ctx + (size_t)SB * SS * SH;            // [4,4,2048,2048]

    char* ws = (char*)d_ws;                              // 16 MB used
    _Float16* Qh = (_Float16*)ws;                        // 4 MB  [b,h,s,d] f16
    _Float16* Kh = (_Float16*)(ws + (size_t)4 * 1024 * 1024);  // 4 MB
    float*    Vf = (float*)(ws + (size_t)8 * 1024 * 1024);     // 8 MB f32

    const size_t attn_elems = (size_t)SB * NH * SS * SS; // 67,108,864 floats
    // each block zeroes 256 threads * 16 floats = 4096 floats
    zero_attn_kernel<<<dim3((unsigned)(attn_elems / 4096)), 256, 0, stream>>>(
        (v4f*)attn);

    qkv_proj_kernel<<<dim3(64, 16, 3), 256, 0, stream>>>(
        x, Wq, Wk, Wv, bq, bk, bv, Qh, Kh, Vf);

    sparse_attn_kernel<<<dim3(128, NH, SB), 256, 0, stream>>>(
        Qh, Kh, Vf, attn, ctx);
}
